// LSTMLanguageModel_10067403341867
// MI455X (gfx1250) — matrix operand order repaired and checked by reading, not compiled, not run on hardware
//
#include <hip/hip_runtime.h>
#include <hip/hip_bf16.h>

typedef __attribute__((ext_vector_type(16))) __bf16         v16bf;
typedef __attribute__((ext_vector_type(8)))  float          v8f;
typedef __attribute__((ext_vector_type(8)))  unsigned short ushort8;

#define VOCAB 30000
#define EMB   100
#define HID   300
#define BSZ   512
#define TT    256
#define NT    19      // 304/16 tiles per gate along H
#define KT_X  4       // EMB padded to 128 -> 4 k-tiles of 32
#define KT_H  10      // HID padded to 320 -> 10 k-tiles of 32
#define HPAD  304
#define VT    1875    // 30000/16

__device__ __forceinline__ unsigned short f2bf(float f) {
    unsigned int u = __float_as_uint(f);
    u += 0x7FFFu + ((u >> 16) & 1u);   // round-to-nearest-even
    return (unsigned short)(u >> 16);
}
__device__ __forceinline__ float sigmoidf(float x) {
    return 1.0f / (1.0f + __expf(-x));
}

// ---- init padded H/C buffers from h0/c0 (pad cols -> 0) -------------------
__global__ void init_hc(const float* __restrict__ h0, const float* __restrict__ c0,
                        float* __restrict__ H, float* __restrict__ C) {
    int i = blockIdx.x * blockDim.x + threadIdx.x;
    if (i >= BSZ * HPAD) return;
    int b = i / HPAD, col = i % HPAD;
    float hv = (col < HID) ? h0[b * HID + col] : 0.0f;
    float cv = (col < HID) ? c0[b * HID + col] : 0.0f;
    H[i] = hv;
    C[i] = cv;
}

// ---- init h A-fragments from h0 -------------------------------------------
// A-frag (16-bit, 16x32): lane l: row m=l%16; kbase=(l/16)*8;
// elem j<8 -> K=kbase+j ; elem j>=8 -> K=kbase+16+(j-8)
__global__ void init_hfrag(const float* __restrict__ h0, unsigned short* __restrict__ hf) {
    int tid = blockIdx.x * blockDim.x + threadIdx.x;   // (bt*10+kt)*32+lane
    if (tid >= 32 * KT_H * 32) return;
    int lane = tid & 31;
    int rec  = tid >> 5;
    int kt   = rec % KT_H;
    int bt   = rec / KT_H;
    int b    = bt * 16 + (lane & 15);
    int kbase = kt * 32 + (lane >> 4) * 8;
    unsigned short* dst = hf + (size_t)tid * 16;
#pragma unroll
    for (int j = 0; j < 16; ++j) {
        int K = kbase + (j & 7) + (j >> 3) * 16;
        float v = (K < HID) ? h0[b * HID + K] : 0.0f;
        dst[j] = f2bf(v);
    }
}

// ---- pack gate weights (W_ih / W_hh) into B-fragments, gate-padded --------
// B-frag (16-bit, 32x16): lane l: col n=l%16; kbase=(l/16)*8; elem e<8 -> K=kbase+e ; e>=8 -> K=kbase+16+(e-8)
// jt = g*19 + nt (N padded 300->304 per gate); zero-fill pad rows/cols.
__global__ void pack_gate_w(const float* __restrict__ W, unsigned short* __restrict__ dst,
                            int Kact, int KT) {
    int tid = blockIdx.x * blockDim.x + threadIdx.x;   // (jt*KT+kt)*32+lane
    if (tid >= 76 * KT * 32) return;
    int lane = tid & 31;
    int rec  = tid >> 5;
    int kt   = rec % KT;
    int jt   = rec / KT;
    int g    = jt / NT, nt = jt % NT;
    int col  = nt * 16 + (lane & 15);
    bool rowvalid = (col < HID);
    int j = g * HID + col;
    int kbase = kt * 32 + (lane >> 4) * 8;
    unsigned short* out = dst + (size_t)tid * 16;
#pragma unroll
    for (int e = 0; e < 16; ++e) {
        int k = kbase + (e & 7) + (e >> 3) * 16;
        float v = (rowvalid && k < Kact) ? W[(size_t)j * Kact + k] : 0.0f;
        out[e] = f2bf(v);
    }
}

// ---- pack W_fc into B-fragments -------------------------------------------
__global__ void pack_fc(const float* __restrict__ W, unsigned short* __restrict__ dst) {
    int tid = blockIdx.x * blockDim.x + threadIdx.x;   // (vt*10+kt)*32+lane
    if (tid >= VT * KT_H * 32) return;
    int lane = tid & 31;
    int rec  = tid >> 5;
    int kt   = rec % KT_H;
    int vt   = rec / KT_H;
    int j    = vt * 16 + (lane & 15);
    int kbase = kt * 32 + (lane >> 4) * 8;
    unsigned short* out = dst + (size_t)tid * 16;
#pragma unroll
    for (int e = 0; e < 16; ++e) {
        int k = kbase + (e & 7) + (e >> 3) * 16;
        float v = (k < HID) ? W[(size_t)j * HID + k] : 0.0f;
        out[e] = f2bf(v);
    }
}

__global__ void bias_combine(const float* __restrict__ bi, const float* __restrict__ bh,
                             float* __restrict__ bc) {
    int i = blockIdx.x * blockDim.x + threadIdx.x;
    if (i < 4 * HID) bc[i] = bi[i] + bh[i];
}

// ---- embedding gather -> A-fragments for all timesteps --------------------
__global__ void embed_pack(const int* __restrict__ x, const float* __restrict__ emb,
                           unsigned short* __restrict__ Xf) {
    int tid = blockIdx.x * blockDim.x + threadIdx.x;   // ((t*32+bt)*4+kt)*32+lane
    if (tid >= TT * 32 * KT_X * 32) return;
    int lane = tid & 31;
    int r    = tid >> 5;
    int kt   = r & 3;  r >>= 2;
    int bt   = r & 31;
    int t    = r >> 5;
    int b    = bt * 16 + (lane & 15);
    int tok  = x[b * TT + t];
    const float* row = emb + (size_t)tok * EMB;
    int kbase = kt * 32 + (lane >> 4) * 8;
    unsigned short* dst = Xf + (size_t)tid * 16;
#pragma unroll
    for (int j = 0; j < 16; ++j) {
        int K = kbase + (j & 7) + (j >> 3) * 16;
        float v = (K < EMB) ? row[K] : 0.0f;
        dst[j] = f2bf(v);
    }
}

// ---- one LSTM timestep: fused dual-GEMM + cell update + h re-pack ---------
// One wave per (bt, nt): computes the 4 gate tiles (i,f,g,o) for a 16x16 h tile.
__global__ void __launch_bounds__(32)
lstm_step(const unsigned short* __restrict__ Xt,    // [32][4][32][16]
          const unsigned short* __restrict__ hIn,   // [32][10][32][16]
          unsigned short* __restrict__ hOut,        // [32][10][32][16]
          const unsigned short* __restrict__ Wih,   // [76][4][32][16]
          const unsigned short* __restrict__ Whh,   // [76][10][32][16]
          const float* __restrict__ bias,           // [1200]
          float* __restrict__ H,                    // [512][304]
          float* __restrict__ C) {                  // [512][304]
    __shared__ float lds[16 * 16];
    const int lane = threadIdx.x;
    const int bt = blockIdx.x / NT;
    const int nt = blockIdx.x % NT;

    v8f acc[4] = {v8f{}, v8f{}, v8f{}, v8f{}};

    // x_t @ W_ih^T (K padded to 128)
#pragma unroll
    for (int kt = 0; kt < KT_X; ++kt) {
        v16bf a = *(const v16bf*)(Xt + (((size_t)bt * KT_X + kt) * 32 + lane) * 16);
#pragma unroll
        for (int g = 0; g < 4; ++g) {
            int jt = g * NT + nt;
            v16bf b = *(const v16bf*)(Wih + (((size_t)jt * KT_X + kt) * 32 + lane) * 16);
            acc[g] = __builtin_amdgcn_wmma_f32_16x16x32_bf16(
                false, a, false, b, (short)0, acc[g], false, false);
        }
    }
    // h_{t-1} @ W_hh^T (K padded to 320)
#pragma unroll
    for (int kt = 0; kt < KT_H; ++kt) {
        v16bf a = *(const v16bf*)(hIn + (((size_t)bt * KT_H + kt) * 32 + lane) * 16);
#pragma unroll
        for (int g = 0; g < 4; ++g) {
            int jt = g * NT + nt;
            v16bf b = *(const v16bf*)(Whh + (((size_t)jt * KT_H + kt) * 32 + lane) * 16);
            acc[g] = __builtin_amdgcn_wmma_f32_16x16x32_bf16(
                false, a, false, b, (short)0, acc[g], false, false);
        }
    }

    const int n    = lane & 15;
    const int half = lane >> 4;
    const int col  = nt * 16 + n;
    const bool valid = (col < HID);
    const float bi = valid ? bias[0 * HID + col] : 0.0f;
    const float bf = valid ? bias[1 * HID + col] : 0.0f;
    const float bg = valid ? bias[2 * HID + col] : 0.0f;
    const float bo = valid ? bias[3 * HID + col] : 0.0f;

    // C/D layout: lane: col n=l%16; elem e: row m = e + 8*(l/16)
#pragma unroll
    for (int e = 0; e < 8; ++e) {
        int m   = e + 8 * half;
        int bix = bt * 16 + m;
        float iv = sigmoidf(acc[0][e] + bi);
        float fv = sigmoidf(acc[1][e] + bf);
        float gv = tanhf(acc[2][e] + bg);
        float ov = sigmoidf(acc[3][e] + bo);
        float cold = valid ? C[(size_t)bix * HPAD + col] : 0.0f;
        float cnew = fv * cold + iv * gv;
        float hv   = ov * tanhf(cnew);
        if (valid) {
            C[(size_t)bix * HPAD + col] = cnew;
            H[(size_t)bix * HPAD + col] = hv;
        }
        lds[m * 16 + n] = hv;   // pad lanes write exact 0 (zero-padded weights)
    }
    __syncthreads();

    // Re-pack this 16-col slice of h into A-frag layout for the next step.
    // This nt covers K-tile nt/2, local K offsets (nt&1)? 16..31 : 0..15.
    int m2 = n;                 // A-frag: lane%16 = row
    int kb = half * 8;          // local col base inside the 16-wide slice
    ushort8 o;
#pragma unroll
    for (int j = 0; j < 8; ++j) o[j] = f2bf(lds[m2 * 16 + kb + j]);
    int kt2 = nt >> 1;
    *(ushort8*)(hOut + (((size_t)bt * KT_H + kt2) * 32 + lane) * 16 + (nt & 1) * 8) = o;
}

// ---- final FC: out = h @ W_fc^T + b_fc ------------------------------------
__global__ void __launch_bounds__(32)
fc_kernel(const unsigned short* __restrict__ hf,   // [32][10][32][16]
          const unsigned short* __restrict__ Wfc,  // [1875][10][32][16]
          const float* __restrict__ bfc, float* __restrict__ out) {
    const int lane = threadIdx.x;
    const int vt = blockIdx.x;
    const int bt = blockIdx.y;
    v8f acc = v8f{};
#pragma unroll
    for (int kt = 0; kt < KT_H; ++kt) {
        v16bf a = *(const v16bf*)(hf  + (((size_t)bt * KT_H + kt) * 32 + lane) * 16);
        v16bf b = *(const v16bf*)(Wfc + (((size_t)vt * KT_H + kt) * 32 + lane) * 16);
        acc = __builtin_amdgcn_wmma_f32_16x16x32_bf16(
            false, a, false, b, (short)0, acc, false, false);
    }
    const int n = lane & 15, half = lane >> 4;
    const float bias = bfc[vt * 16 + n];
#pragma unroll
    for (int e = 0; e < 8; ++e) {
        int m = e + 8 * half;
        out[(size_t)(bt * 16 + m) * VOCAB + vt * 16 + n] = acc[e] + bias;
    }
}

// ---- copy final h, c into output tail -------------------------------------
__global__ void copy_hc(const float* __restrict__ H, const float* __restrict__ C,
                        float* __restrict__ out) {
    int i = blockIdx.x * blockDim.x + threadIdx.x;
    if (i >= BSZ * HID) return;
    int b = i / HID, nn = i % HID;
    out[(size_t)BSZ * VOCAB + i]             = H[(size_t)b * HPAD + nn];
    out[(size_t)BSZ * VOCAB + BSZ * HID + i] = C[(size_t)b * HPAD + nn];
}

extern "C" void kernel_launch(void* const* d_in, const int* in_sizes, int n_in,
                              void* d_out, int out_size, void* d_ws, size_t ws_size,
                              hipStream_t stream) {
    const int*   x    = (const int*)  d_in[0];
    const float* h0   = (const float*)d_in[1];
    const float* c0   = (const float*)d_in[2];
    const float* emb  = (const float*)d_in[3];
    const float* W_ih = (const float*)d_in[4];
    const float* W_hh = (const float*)d_in[5];
    const float* b_ih = (const float*)d_in[6];
    const float* b_hh = (const float*)d_in[7];
    const float* W_fc = (const float*)d_in[8];
    const float* b_fc = (const float*)d_in[9];
    float* out = (float*)d_out;

    size_t off = 0;
    char* base = (char*)d_ws;
    auto alloc = [&](size_t bytes) -> void* {
        off = (off + 255) & ~(size_t)255;
        void* p = base + off;
        off += bytes;
        return p;
    };
    unsigned short* Xfrag = (unsigned short*)alloc((size_t)TT * 32 * KT_X * 32 * 16 * 2);
    unsigned short* WihP  = (unsigned short*)alloc((size_t)76 * KT_X * 32 * 16 * 2);
    unsigned short* WhhP  = (unsigned short*)alloc((size_t)76 * KT_H * 32 * 16 * 2);
    unsigned short* WfcP  = (unsigned short*)alloc((size_t)VT * KT_H * 32 * 16 * 2);
    float*          bcomb = (float*)alloc(4 * HID * sizeof(float));
    unsigned short* hfA   = (unsigned short*)alloc((size_t)32 * KT_H * 32 * 16 * 2);
    unsigned short* hfB   = (unsigned short*)alloc((size_t)32 * KT_H * 32 * 16 * 2);
    float*          Hbuf  = (float*)alloc((size_t)BSZ * HPAD * sizeof(float));
    float*          Cbuf  = (float*)alloc((size_t)BSZ * HPAD * sizeof(float));

    init_hc<<<(BSZ * HPAD + 255) / 256, 256, 0, stream>>>(h0, c0, Hbuf, Cbuf);
    init_hfrag<<<(32 * KT_H * 32 + 255) / 256, 256, 0, stream>>>(h0, hfA);
    pack_gate_w<<<(76 * KT_X * 32 + 255) / 256, 256, 0, stream>>>(W_ih, WihP, EMB, KT_X);
    pack_gate_w<<<(76 * KT_H * 32 + 255) / 256, 256, 0, stream>>>(W_hh, WhhP, HID, KT_H);
    pack_fc<<<(VT * KT_H * 32 + 255) / 256, 256, 0, stream>>>(W_fc, WfcP);
    bias_combine<<<(4 * HID + 255) / 256, 256, 0, stream>>>(b_ih, b_hh, bcomb);
    embed_pack<<<(TT * 32 * KT_X * 32) / 256, 256, 0, stream>>>(x, emb, Xfrag);

    unsigned short* hin = hfA;
    unsigned short* hout = hfB;
    for (int t = 0; t < TT; ++t) {
        lstm_step<<<32 * NT, 32, 0, stream>>>(
            Xfrag + (size_t)t * 32 * KT_X * 32 * 16,
            hin, hout, WihP, WhhP, bcomb, Hbuf, Cbuf);
        unsigned short* tmp = hin; hin = hout; hout = tmp;
    }

    fc_kernel<<<dim3(VT, 32), 32, 0, stream>>>(hin, WfcP, b_fc, out);
    copy_hc<<<(BSZ * HID + 255) / 256, 256, 0, stream>>>(Hbuf, Cbuf, out);
    (void)in_sizes; (void)n_in; (void)out_size; (void)ws_size;
}
